// ViT_51943334478259
// MI455X (gfx1250) — matrix operand order repaired and checked by reading, not compiled, not run on hardware
//
#include <hip/hip_runtime.h>
#include <hip/hip_bf16.h>

// ---------------------------------------------------------------------------
// ViT forward for gfx1250 (MI455X): f16 WMMA GEMMs with f32 accumulation.
// Double-buffered LDS tiles, branch-free clamped staging, vectorized loads,
// fragment preloading so the 4 WMMAs issue back-to-back behind one dscnt wait.
// ---------------------------------------------------------------------------

typedef __attribute__((ext_vector_type(16))) _Float16 v16h;
typedef __attribute__((ext_vector_type(8)))  _Float16 v8h;
typedef __attribute__((ext_vector_type(8)))  float    v8f;

#define BDIM 256
static const int kB  = 64;
static const int kS  = 197;   // 14*14 + 1
static const int kD  = 768;
static const int kNH = 12;
static const int kDH = 64;
static const int kF  = 3072;
static const int kNC = 1000;
static const int kL  = 12;
static const int kMT = kB * kS;          // 12608 token rows
static const int kNP = 196;              // patches per image

__device__ __forceinline__ v16h mk16(v8h lo, v8h hi) {
  v16h r;
#pragma unroll
  for (int i = 0; i < 8; ++i) { r[i] = lo[i]; r[i + 8] = hi[i]; }
  return r;
}

// ---------------------------------------------------------------------------
// Generic batched WMMA GEMM:
//   C[z] = alpha * (A[z] (MxK) * W[z] (KxN)) + bias[z] (+ReLU) (+R[z])
// Block tile: 128(M) x 64(N), K-step 32, 8 waves each computing 16x64.
// z-batch offsets: off = (z / zdiv)*s?1 + (z % zdiv)*s?2  (element units).
// transW: W stored as (N x K) row-major (i.e. we need W^T), ld = ldw.
// ---------------------------------------------------------------------------
__global__ __launch_bounds__(BDIM)
void k_wmma_gemm(const float* __restrict__ A,  long long sA1, long long sA2, int lda,
                 const float* __restrict__ W,  long long sW1, long long sW2, int ldw, int transW,
                 const float* __restrict__ bias, long long sB1, long long sB2,
                 const float* __restrict__ R,  long long sR1, long long sR2, int ldr,
                 float* __restrict__ C,        long long sC1, long long sC2, int ldc,
                 int M, int N, int K, int zdiv, float alpha, int relu)
{
  // double-buffered tiles: A 128x32 halves (+8 pad), W^T 64x32 halves (+8 pad)
  __shared__ alignas(16) _Float16 As[2][128 * 40];
  __shared__ alignas(16) _Float16 Bs[2][64 * 40];

  const int z  = blockIdx.z;
  const int z1 = z / zdiv, z2 = z % zdiv;
  A += z1 * sA1 + z2 * sA2;
  W += z1 * sW1 + z2 * sW2;
  if (bias) bias += z1 * sB1 + z2 * sB2;
  if (R)    R    += z1 * sR1 + z2 * sR2;
  C += z1 * sC1 + z2 * sC2;

  const int blockM = blockIdx.y * 128;
  const int blockN = blockIdx.x * 64;

  const int tid  = threadIdx.x;
  const int lane = tid & 31;
  const int wave = tid >> 5;        // 0..7 -> 16-row M sub-tile
  const int l15  = lane & 15;
  const int lhi  = lane >> 4;       // 0/1

  // --- staging assignments (per thread) ---
  const int  arow = tid >> 1;              // 0..127
  const int  acb  = (tid & 1) * 16;        // 0 / 16
  const int  amg  = blockM + arow;
  const bool arok = (amg < M);
  const float* aptr = A + (size_t)(arok ? amg : 0) * lda;   // row-clamped, always valid
  const bool aVec = ((lda & 3) == 0);

  const int  wn   = tid >> 2;              // 0..63
  const int  wkb  = (tid & 3) * 8;         // 0/8/16/24
  const int  wng  = blockN + wn;
  const bool wnok = (wng < N);
  const int  wngc = wnok ? wng : 0;
  const bool wVec = (transW && ((ldw & 3) == 0));

  float ar[16];
  float wr[8];

  auto loadTiles = [&](int k0) {
    // ---- A tile: 16 contiguous K values per thread ----
    if (aVec && arok && (k0 + 32) <= K) {
      const float4* p = (const float4*)(aptr + k0 + acb);
#pragma unroll
      for (int j = 0; j < 4; ++j) {
        const float4 t = p[j];
        ar[4 * j + 0] = t.x; ar[4 * j + 1] = t.y;
        ar[4 * j + 2] = t.z; ar[4 * j + 3] = t.w;
      }
    } else {
#pragma unroll
      for (int j = 0; j < 16; ++j) {
        const int kk  = k0 + acb + j;
        const int kkc = (kk < K) ? kk : (K - 1);       // clamped -> speculatable
        const float vv = aptr[kkc];
        ar[j] = (arok && kk < K) ? vv : 0.0f;
      }
    }
    // prefetch two K-steps beyond the one we are staging
    if (arok && (k0 + 64) < K) __builtin_prefetch(aptr + k0 + 64 + acb, 0, 3);

    // ---- W tile: 8 K values per thread, stored transposed Bs[n][kk] ----
    if (wVec && wnok && (k0 + 32) <= K) {
      const float4* p = (const float4*)(W + (size_t)wngc * ldw + k0 + wkb);
      const float4 t0 = p[0], t1 = p[1];
      wr[0] = t0.x; wr[1] = t0.y; wr[2] = t0.z; wr[3] = t0.w;
      wr[4] = t1.x; wr[5] = t1.y; wr[6] = t1.z; wr[7] = t1.w;
    } else {
#pragma unroll
      for (int j = 0; j < 8; ++j) {
        const int kk  = k0 + wkb + j;
        const int kkc = (kk < K) ? kk : (K - 1);
        const float vv = transW ? W[(size_t)wngc * ldw + kkc]
                                : W[(size_t)kkc * ldw + wngc];
        wr[j] = (wnok && kk < K) ? vv : 0.0f;
      }
    }
  };

  auto storeTiles = [&](int buf) {
    v8h h0, h1, w0;
#pragma unroll
    for (int i = 0; i < 8; ++i) {
      h0[i] = (_Float16)ar[i];
      h1[i] = (_Float16)ar[8 + i];
      w0[i] = (_Float16)wr[i];
    }
    *(v8h*)&As[buf][arow * 40 + acb]     = h0;
    *(v8h*)&As[buf][arow * 40 + acb + 8] = h1;
    *(v8h*)&Bs[buf][wn * 40 + wkb]       = w0;
  };

  v8f acc[4] = {};
  const int nsteps = (K + 31) / 32;

  loadTiles(0);
  storeTiles(0);
  __syncthreads();

  for (int s = 0; s < nsteps; ++s) {
    const int  buf     = s & 1;
    const bool hasNext = (s + 1) < nsteps;

    if (hasNext) loadTiles((s + 1) * 32);   // global loads overlap WMMA below

    // ---- preload ALL fragments first so WMMAs can issue back-to-back ----
    // A fragment (16x32 ISA layout): lanes 0-15 K {0-7,16-23}, lanes 16-31 K {8-15,24-31}
    const _Float16* arp = &As[buf][(wave * 16 + l15) * 40];
    const v8h alo = *(const v8h*)(arp + lhi * 8);
    const v8h ahi = *(const v8h*)(arp + 16 + lhi * 8);

    // B fragments (32x16): lane -> column n, K {0-7,16-23} / {8-15,24-31} like A
    v16h bf[4];
#pragma unroll
    for (int ns = 0; ns < 4; ++ns) {
      const _Float16* brp = &Bs[buf][(ns * 16 + l15) * 40 + lhi * 8];
      bf[ns] = mk16(*(const v8h*)brp, *(const v8h*)(brp + 16));
    }
    const v16h af = mk16(alo, ahi);

#pragma unroll
    for (int ns = 0; ns < 4; ++ns) {
      acc[ns] = __builtin_amdgcn_wmma_f32_16x16x32_f16(
          false, af, false, bf[ns], (short)0, acc[ns], false, false);
    }

    if (hasNext) {
      storeTiles(buf ^ 1);   // write other buffer; readers of it synced last iter
      __syncthreads();
    }
  }

  // --- store: C/D layout: VGPR r -> M = r + lhi*8, N = l15 ---
  const int mbase = blockM + wave * 16 + lhi * 8;
#pragma unroll
  for (int ns = 0; ns < 4; ++ns) {
    const int ng = blockN + ns * 16 + l15;
    if (ng >= N) continue;
    const float bb = bias ? bias[ng] : 0.0f;
#pragma unroll
    for (int r = 0; r < 8; ++r) {
      const int mg = mbase + r;
      if (mg >= M) continue;
      float val = alpha * acc[ns][r] + bb;
      if (relu) val = fmaxf(val, 0.0f);
      if (R) val += R[(size_t)mg * ldr + ng];
      C[(size_t)mg * ldc + ng] = val;
    }
  }
}

// ---------------------------------------------------------------------------
// Patchify: images (B,3,224,224) -> Xp (B*196, 768), d = ((i*16 + j)*3 + c)
// ---------------------------------------------------------------------------
__global__ __launch_bounds__(BDIM)
void k_patchify(const float* __restrict__ img, float* __restrict__ Xp)
{
  const int total = kB * kNP * 768;
  int idx = blockIdx.x * BDIM + threadIdx.x;
  if (idx >= total) return;
  const int d  = idx % 768;
  const int t  = idx / 768;
  const int p  = t % kNP;
  const int b  = t / kNP;
  const int c  = d % 3;
  const int ij = d / 3;
  const int j  = ij % 16;
  const int i  = ij / 16;
  const int px = p % 14, py = p / 14;
  const int row = py * 16 + i, col = px * 16 + j;
  Xp[idx] = img[(((size_t)b * 3 + c) * 224 + row) * 224 + col];
}

// x[b,s,:] = (s==0 ? cls : emb[b, s-1]) + pos[s]
__global__ __launch_bounds__(BDIM)
void k_assemble(const float* __restrict__ emb, const float* __restrict__ cls,
                const float* __restrict__ pos, float* __restrict__ x)
{
  const int total = kMT * kD;
  int idx = blockIdx.x * BDIM + threadIdx.x;
  if (idx >= total) return;
  const int d = idx % kD;
  const int t = idx / kD;
  const int s = t % kS;
  const int b = t / kS;
  const float v = (s == 0) ? cls[d] : emb[((size_t)b * kNP + (s - 1)) * kD + d];
  x[idx] = v + pos[(size_t)s * kD + d];
}

// Row LayerNorm: one block per row of length Dn
__global__ __launch_bounds__(BDIM)
void k_ln(const float* __restrict__ x, const float* __restrict__ g,
          const float* __restrict__ b, float* __restrict__ out, int Dn)
{
  __shared__ float red[BDIM];
  const int row = blockIdx.x;
  const int tid = threadIdx.x;
  const float* xr = x + (size_t)row * Dn;

  float s = 0.0f;
  for (int i = tid; i < Dn; i += BDIM) s += xr[i];
  red[tid] = s; __syncthreads();
  for (int o = BDIM / 2; o > 0; o >>= 1) { if (tid < o) red[tid] += red[tid + o]; __syncthreads(); }
  const float mean = red[0] / Dn;
  __syncthreads();

  float v = 0.0f;
  for (int i = tid; i < Dn; i += BDIM) { const float dd = xr[i] - mean; v += dd * dd; }
  red[tid] = v; __syncthreads();
  for (int o = BDIM / 2; o > 0; o >>= 1) { if (tid < o) red[tid] += red[tid + o]; __syncthreads(); }
  const float rstd = rsqrtf(red[0] / Dn + 1e-5f);
  __syncthreads();

  float* orow = out + (size_t)row * Dn;
  for (int i = tid; i < Dn; i += BDIM)
    orow[i] = (xr[i] - mean) * rstd * g[i] + b[i];
}

// Row softmax in place, row length Sn
__global__ __launch_bounds__(BDIM)
void k_softmax(float* __restrict__ sc, int Sn)
{
  __shared__ float red[BDIM];
  float* p = sc + (size_t)blockIdx.x * Sn;
  const int tid = threadIdx.x;

  float m = -1e30f;
  for (int i = tid; i < Sn; i += BDIM) m = fmaxf(m, p[i]);
  red[tid] = m; __syncthreads();
  for (int o = BDIM / 2; o > 0; o >>= 1) { if (tid < o) red[tid] = fmaxf(red[tid], red[tid + o]); __syncthreads(); }
  m = red[0]; __syncthreads();

  float s = 0.0f;
  for (int i = tid; i < Sn; i += BDIM) { const float e = __expf(p[i] - m); p[i] = e; s += e; }
  red[tid] = s; __syncthreads();
  for (int o = BDIM / 2; o > 0; o >>= 1) { if (tid < o) red[tid] += red[tid + o]; __syncthreads(); }
  const float inv = 1.0f / red[0];
  __syncthreads();
  for (int i = tid; i < Sn; i += BDIM) p[i] *= inv;
}

// cls rows of x -> out (B x D)
__global__ __launch_bounds__(BDIM)
void k_cls(const float* __restrict__ x, float* __restrict__ out)
{
  const int total = kB * kD;
  int idx = blockIdx.x * BDIM + threadIdx.x;
  if (idx >= total) return;
  const int d = idx % kD;
  const int b = idx / kD;
  out[idx] = x[(size_t)b * kS * kD + d];
}

// ---------------------------------------------------------------------------
// Host orchestration
// ---------------------------------------------------------------------------
static inline void gemm(hipStream_t st,
                        const float* A, long long sA1, long long sA2, int lda,
                        const float* W, long long sW1, long long sW2, int ldw, int transW,
                        const float* bias, long long sB1, long long sB2,
                        const float* R, long long sR1, long long sR2, int ldr,
                        float* C, long long sC1, long long sC2, int ldc,
                        int M, int N, int K, int nz, int zdiv, float alpha, int relu)
{
  dim3 grid((N + 63) / 64, (M + 127) / 128, nz);
  k_wmma_gemm<<<grid, BDIM, 0, st>>>(A, sA1, sA2, lda, W, sW1, sW2, ldw, transW,
                                     bias, sB1, sB2, R, sR1, sR2, ldr,
                                     C, sC1, sC2, ldc, M, N, K, zdiv, alpha, relu);
}

extern "C" void kernel_launch(void* const* d_in, const int* in_sizes, int n_in,
                              void* d_out, int out_size, void* d_ws, size_t ws_size,
                              hipStream_t stream)
{
  const float* images = (const float*)d_in[0];
  const float* emb_w  = (const float*)d_in[1];
  const float* emb_b  = (const float*)d_in[2];
  const float* cls_t  = (const float*)d_in[3];
  const float* pos    = (const float*)d_in[4];
  const float* n1_g   = (const float*)d_in[5];
  const float* n1_b   = (const float*)d_in[6];
  const float* wq     = (const float*)d_in[7];
  const float* bq     = (const float*)d_in[8];
  const float* wk     = (const float*)d_in[9];
  const float* bk     = (const float*)d_in[10];
  const float* wv     = (const float*)d_in[11];
  const float* bv     = (const float*)d_in[12];
  const float* n2_g   = (const float*)d_in[13];
  const float* n2_b   = (const float*)d_in[14];
  const float* w1     = (const float*)d_in[15];
  const float* b1     = (const float*)d_in[16];
  const float* w2     = (const float*)d_in[17];
  const float* b2     = (const float*)d_in[18];
  const float* hw     = (const float*)d_in[19];
  const float* hb     = (const float*)d_in[20];
  float* out = (float*)d_out;

  // workspace layout (f32): x | h | q | k | v | big(scores / mlp-hidden / patches)
  float* ws = (float*)d_ws;
  const size_t TD = (size_t)kMT * kD;             // 12608*768
  float* x    = ws;
  float* h    = x + TD;
  float* q    = h + TD;
  float* kbuf = q + TD;
  float* v    = kbuf + TD;
  float* big  = v + TD;                            // up to 12608*3072 floats

  const long long SS  = (long long)kS * kS;        // 38809
  const float scale = 0.125f;                      // 1/sqrt(64)

  // 1) patchify -> big
  {
    const int total = kB * kNP * 768;
    k_patchify<<<(total + BDIM - 1) / BDIM, BDIM, 0, stream>>>(images, big);
  }
  // 2) patch embed: big (12544x768) @ emb_w + emb_b -> h
  gemm(stream, big, 0, 0, kD, emb_w, 0, 0, kD, 0, emb_b, 0, 0,
       nullptr, 0, 0, 0, h, 0, 0, kD, kB * kNP, kD, kD, 1, 1, 1.0f, 0);
  // 3) assemble x = [cls; h] + pos
  {
    const int total = kMT * kD;
    k_assemble<<<(total + BDIM - 1) / BDIM, BDIM, 0, stream>>>(h, cls_t, pos, x);
  }

  for (int l = 0; l < kL; ++l) {
    const float* wql = wq + (size_t)l * kNH * kDH * kDH;
    const float* wkl = wk + (size_t)l * kNH * kDH * kDH;
    const float* wvl = wv + (size_t)l * kNH * kDH * kDH;
    const float* bql = bq + (size_t)l * kD;
    const float* bkl = bk + (size_t)l * kD;
    const float* bvl = bv + (size_t)l * kD;

    // LN1: x -> h
    k_ln<<<kMT, BDIM, 0, stream>>>(x, n1_g + (size_t)l * kD, n1_b + (size_t)l * kD, h, kD);

    // per-head QKV: z = head (zdiv = 12, z1 = 0)
    gemm(stream, h, 0, kDH, kD, wql, 0, kDH * kDH, kDH, 0, bql, 0, kDH,
         nullptr, 0, 0, 0, q, 0, kDH, kD, kMT, kDH, kDH, kNH, kNH, 1.0f, 0);
    gemm(stream, h, 0, kDH, kD, wkl, 0, kDH * kDH, kDH, 0, bkl, 0, kDH,
         nullptr, 0, 0, 0, kbuf, 0, kDH, kD, kMT, kDH, kDH, kNH, kNH, 1.0f, 0);
    gemm(stream, h, 0, kDH, kD, wvl, 0, kDH * kDH, kDH, 0, bvl, 0, kDH,
         nullptr, 0, 0, 0, v, 0, kDH, kD, kMT, kDH, kDH, kNH, kNH, 1.0f, 0);

    // scores = scale * q @ k^T : z = b*NH + h
    gemm(stream, q, (long long)kS * kD, kDH, kD,
         kbuf, (long long)kS * kD, kDH, kD, /*transW=*/1,
         nullptr, 0, 0, nullptr, 0, 0, 0,
         big, kNH * SS, SS, kS,
         kS, kS, kDH, kB * kNH, kNH, scale, 0);

    // softmax over rows of scores
    k_softmax<<<kB * kNH * kS, BDIM, 0, stream>>>(big, kS);

    // x += att @ v   (residual accumulate straight into x)
    gemm(stream, big, kNH * SS, SS, kS,
         v, (long long)kS * kD, kDH, kD, 0,
         nullptr, 0, 0,
         x, (long long)kS * kD, kDH, kD,
         x, (long long)kS * kD, kDH, kD,
         kS, kDH, kS, kB * kNH, kNH, 1.0f, 0);

    // LN2: x -> h
    k_ln<<<kMT, BDIM, 0, stream>>>(x, n2_g + (size_t)l * kD, n2_b + (size_t)l * kD, h, kD);

    // MLP1: big = relu(h @ w1 + b1)
    gemm(stream, h, 0, 0, kD, w1 + (size_t)l * kD * kF, 0, 0, kF, 0,
         b1 + (size_t)l * kF, 0, 0, nullptr, 0, 0, 0,
         big, 0, 0, kF, kMT, kF, kD, 1, 1, 1.0f, 1);

    // MLP2: x = x + big @ w2 + b2
    gemm(stream, big, 0, 0, kF, w2 + (size_t)l * kF * kD, 0, 0, kD, 0,
         b2 + (size_t)l * kD, 0, 0, x, 0, 0, kD,
         x, 0, 0, kD, kMT, kD, kF, 1, 1, 1.0f, 0);
  }

  // head: logits = cls_rows @ head_w + head_b   (A rows strided by S*D)
  gemm(stream, x, 0, 0, kS * kD, hw, 0, 0, kNC, 0, hb, 0, 0,
       nullptr, 0, 0, 0, out, 0, 0, kNC, kB, kNC, kD, 1, 1, 1.0f, 0);

  // cls output after the 64*1000 logits
  {
    const int total = kB * kD;
    k_cls<<<(total + BDIM - 1) / BDIM, BDIM, 0, stream>>>(x, out + (size_t)kB * kNC);
  }
}
